// RNNClass_23295902613557
// MI455X (gfx1250) — matrix operand order repaired and checked by reading, not compiled, not run on hardware
//
#include <hip/hip_runtime.h>
#include <hip/hip_bf16.h>
#include <math.h>

// Problem constants (reference: B=16, T=4096, D=1024)
#define B_  16
#define T_  4096
#define D_  1024

typedef __attribute__((ext_vector_type(16))) __bf16 v16bf;
typedef __attribute__((ext_vector_type(8)))  __bf16 v8bf;
typedef __attribute__((ext_vector_type(4)))  __bf16 v4bf;
typedef __attribute__((ext_vector_type(8)))  float  v8f;
typedef __attribute__((ext_vector_type(4)))  int    v4i;

#if defined(__HIP_DEVICE_COMPILE__) && defined(__has_builtin)
#if __has_builtin(__builtin_amdgcn_global_load_async_to_lds_b128)
#define HAVE_ASYNC_LDS 1
#endif
#endif

// generic->AS pointers via integer round trip:
//  - LDS: low 32 bits of a flat LDS address are the LDS byte offset (ISA 10.2)
//  - global: flat address == global address
// Builtin signature wants int4-typed pointers (per clang diagnostic).
#define TO_GLB(p) ((__attribute__((address_space(1))) v4i*)(size_t)(p))
#define TO_LDS(p) ((__attribute__((address_space(3))) v4i*)(unsigned)(size_t)(p))

// copy 16 bytes global -> LDS (async on CDNA5 if available)
static __device__ __forceinline__ void copy16_g2l(const __bf16* g, __bf16* l) {
#ifdef HAVE_ASYNC_LDS
    __builtin_amdgcn_global_load_async_to_lds_b128(TO_GLB(g), TO_LDS(l), 0, 0);
#else
    *(v8bf*)l = *(const v8bf*)g;
#endif
}

static __device__ __forceinline__ void async_wait0() {
#ifdef HAVE_ASYNC_LDS
#if __has_builtin(__builtin_amdgcn_s_wait_asynccnt)
    __builtin_amdgcn_s_wait_asynccnt(0);
#else
    asm volatile("s_wait_asynccnt 0" ::: "memory");
#endif
#endif
}

static __device__ __forceinline__ v16bf cat8(v8bf lo, v8bf hi) {
    return __builtin_shufflevector(lo, hi, 0,1,2,3,4,5,6,7,8,9,10,11,12,13,14,15);
}

static __device__ __forceinline__ v8f wmma_bf16(v16bf a, v16bf b, v8f c) {
    // D(16x16,f32) = A(16x32,bf16) x B(32x16,bf16) + C
    return __builtin_amdgcn_wmma_f32_16x16x32_bf16(
        /*neg_a=*/false, a, /*neg_b=*/false, b,
        /*c_mod=*/(short)0, c, /*reuse_a=*/false, /*reuse_b=*/false);
}

// ---------------------------------------------------------------------------
// k_init: f32 -> bf16 conversions of Wih, Whh, h0 + zero barrier counters.
// ---------------------------------------------------------------------------
__global__ __launch_bounds__(256) void k_init(
    const float* __restrict__ Wih, const float* __restrict__ Whh,
    const float* __restrict__ h0,
    __bf16* __restrict__ WihBf, __bf16* __restrict__ WhhBf,
    __bf16* __restrict__ Hbf, unsigned* __restrict__ bar)
{
    unsigned i = blockIdx.x * 256u + threadIdx.x;
    if (i < 1048576u)                 WihBf[i]            = (__bf16)Wih[i];
    else if (i < 2097152u)            WhhBf[i - 1048576u] = (__bf16)Whh[i - 1048576u];
    else if (i < 2097152u + 16384u)   Hbf[i - 2097152u]   = (__bf16)h0[i - 2097152u];
    if (i == 0) { bar[0] = 0u; bar[1] = 0u; }
}

// ---------------------------------------------------------------------------
// k_xproj: xproj[t][b][n] = sum_d x[b][t][d]*Wih[n][d] + bih[n]
// M = t*16+b (65536), N = 1024, K = 1024.
// Block tile 64(M) x 128(N); 8 waves in a 4(M) x 2(N) grid; each wave owns a
// 16x64 strip = 4 accumulators, reusing its A fragment 4x per K-chunk of 32.
// ---------------------------------------------------------------------------
__global__ __launch_bounds__(256) void k_xproj(
    const float* __restrict__ x, const __bf16* __restrict__ WihBf,
    const float* __restrict__ bih, float* __restrict__ xproj)
{
    __shared__ __bf16 ldsX[64 * 40];    // 64 M-rows x 32 K (stride 40 halves, 16B-aligned rows)
    __shared__ __bf16 ldsW[128 * 40];   // 128 N-rows x 32 K

    const int tid  = threadIdx.x;
    const int lane = tid & 31;
    const int wv   = tid >> 5;           // 0..7
    const int g    = lane >> 4;
    const int lr   = lane & 15;

    const int Nbase = (blockIdx.x & 7) * 128;
    const int Mbase = (blockIdx.x >> 3) * 64;
    const int mi = wv >> 1;              // 0..3 : wave M sub-tile
    const int ni = wv & 1;               // 0..1 : wave N half (64 cols)
    const int M0 = Mbase + mi * 16;

    // X staging: 64 rows x 32 f32, one thread = 8 cols of one row
    const int sxr = tid >> 2;            // 0..63
    const int sxc = (tid & 3) * 8;       // 0,8,16,24
    const int m_s = Mbase + sxr;         // m = t*16 + b
    const float* xrow = x + (size_t)(m_s & 15) * (size_t)(T_ * D_)
                          + (size_t)(m_s >> 4) * D_;

    v8f acc0 = {}, acc1 = {}, acc2 = {}, acc3 = {};

    for (int k0 = 0; k0 < D_; k0 += 32) {
        __syncthreads();                 // protect LDS from previous iteration
        // stage X tile (f32 -> bf16, ds_store)
        float4 f0 = *(const float4*)(xrow + k0 + sxc);
        float4 f1 = *(const float4*)(xrow + k0 + sxc + 4);
        v8bf xb;
        xb[0]=(__bf16)f0.x; xb[1]=(__bf16)f0.y; xb[2]=(__bf16)f0.z; xb[3]=(__bf16)f0.w;
        xb[4]=(__bf16)f1.x; xb[5]=(__bf16)f1.y; xb[6]=(__bf16)f1.z; xb[7]=(__bf16)f1.w;
        *(v8bf*)&ldsX[sxr * 40 + sxc] = xb;
        // stage W tile: 128 rows x 64B = 512 x 16B transfers, 2 per thread (async)
#pragma unroll
        for (int j = 0; j < 2; ++j) {
            int idx = tid + j * 256;     // 0..511
            int row = idx >> 2;
            int seg = (idx & 3) * 8;     // halves
            copy16_g2l(WihBf + (size_t)(Nbase + row) * D_ + k0 + seg,
                       &ldsW[row * 40 + seg]);
        }
        async_wait0();
        __syncthreads();

        // A fragment: lane row M=l&15; halves h -> K=16g+h
        const __bf16* ar = &ldsX[(mi * 16 + lr) * 40];
        v16bf a = cat8(*(const v8bf*)(ar + 16 * g),
                       *(const v8bf*)(ar + 16 * g + 8));
        // B fragments: lane col N=l&15; halves h -> K=16g+h (B[k][n] = Wih[n][k])
#pragma unroll
        for (int j = 0; j < 4; ++j) {
            const __bf16* br = &ldsW[(ni * 64 + j * 16 + lr) * 40 + 16 * g];
            v16bf b = cat8(*(const v8bf*)br, *(const v8bf*)(br + 8));
            if (j == 0) acc0 = wmma_bf16(a, b, acc0);
            if (j == 1) acc1 = wmma_bf16(a, b, acc1);
            if (j == 2) acc2 = wmma_bf16(a, b, acc2);
            if (j == 3) acc3 = wmma_bf16(a, b, acc3);
        }
    }

    // epilogue: C layout M = v + 8g, N = l&15
#pragma unroll
    for (int j = 0; j < 4; ++j) {
        v8f acc = (j == 0) ? acc0 : (j == 1) ? acc1 : (j == 2) ? acc2 : acc3;
        const int n = Nbase + ni * 64 + j * 16 + lr;
        const float bn = bih[n];
#pragma unroll
        for (int v = 0; v < 8; ++v) {
            int m = M0 + v + 8 * g;
            int t = m >> 4, b = m & 15;
            xproj[(size_t)t * (B_ * D_) + b * D_ + n] = acc[v] + bn;
        }
    }
}

// ---------------------------------------------------------------------------
// k_scan: persistent kernel, 8 blocks x 8 waves = 64 waves = 64 N-tiles.
// Per step: h_new = tanh(xproj_t + h @ Whh^T + bhh); atomic grid barrier.
// ---------------------------------------------------------------------------
__global__ __launch_bounds__(256) void k_scan(
    const __bf16* __restrict__ WhhBf, const float* __restrict__ xproj,
    const float* __restrict__ bhh, __bf16* __restrict__ Hbf,
    float* __restrict__ out, unsigned* __restrict__ bar)
{
    __shared__ __bf16 ldsH[B_ * D_];     // 16 x 1024 bf16 = 32 KB

    const int tid  = threadIdx.x;
    const int lane = tid & 31;
    const int wv   = tid >> 5;
    const int g    = lane >> 4;
    const int lr   = lane & 15;

    const int nt = blockIdx.x * 8 + wv;  // 0..63
    const int N0 = nt * 16;
    const int n  = N0 + lr;
    const float bn = bhh[n];
    const __bf16* wrow = WhhBf + (size_t)n * D_;   // B[k][n] = Whh[n][k]

    for (int t = 0; t < T_; ++t) {
        const __bf16* hc = Hbf + (t & 1) * (B_ * D_);
        // stage full h (16x1024 bf16 = 2048 x 16B) into LDS via async copies
#pragma unroll
        for (int i = 0; i < 8; ++i) {
            int e = (tid + i * 256) * 8;          // half index, 16B granules
            copy16_g2l(hc + e, &ldsH[e]);
        }
        const float* xp = xproj + (size_t)t * (B_ * D_);
        __builtin_prefetch(xp + tid * 64, 0, 1);  // pull xproj_t toward L2
        async_wait0();
        __syncthreads();

        v8f acc = {};
#pragma unroll 4
        for (int k0 = 0; k0 < D_; k0 += 32) {
            const __bf16* ar = &ldsH[lr * D_ + k0];
            v16bf a = cat8(*(const v8bf*)(ar + 16 * g),
                           *(const v8bf*)(ar + 16 * g + 8));
            v16bf b = *(const v16bf*)(wrow + k0 + 16 * g);  // 32B-aligned
            acc = wmma_bf16(a, b, acc);
        }

        __bf16* hn = Hbf + ((t + 1) & 1) * (B_ * D_);
#pragma unroll
        for (int v = 0; v < 8; ++v) {
            int m = v + 8 * g;                    // batch index
            float val = acc[v] + xp[m * D_ + n] + bn;
            float hv  = tanhf(val);
            hn[m * D_ + n] = (__bf16)hv;
            if (t == T_ - 1) out[m * D_ + n] = hv;
        }

        // ---- grid barrier (8 blocks, sense via generation counter) ----
        __syncthreads();                          // also guards ldsH reuse
        if (tid == 0) {
            __threadfence();
            unsigned gen0 = __hip_atomic_load(&bar[1], __ATOMIC_ACQUIRE,
                                              __HIP_MEMORY_SCOPE_AGENT);
            unsigned arrived = __hip_atomic_fetch_add(&bar[0], 1u, __ATOMIC_ACQ_REL,
                                                      __HIP_MEMORY_SCOPE_AGENT);
            if (arrived == gridDim.x - 1) {
                __hip_atomic_store(&bar[0], 0u, __ATOMIC_RELAXED,
                                   __HIP_MEMORY_SCOPE_AGENT);
                __hip_atomic_fetch_add(&bar[1], 1u, __ATOMIC_RELEASE,
                                       __HIP_MEMORY_SCOPE_AGENT);
            } else {
                while (__hip_atomic_load(&bar[1], __ATOMIC_ACQUIRE,
                                         __HIP_MEMORY_SCOPE_AGENT) == gen0)
                    __builtin_amdgcn_s_sleep(2);
            }
        }
        __syncthreads();
    }
}

// ---------------------------------------------------------------------------
// Workspace layout (bytes):
//   [0, 256MiB)            xproj f32 [T][B][D]
//   +268435456 : 2 MiB     Whh bf16
//   +270532608 : 2 MiB     Wih bf16
//   +272629760 : 64 KiB    h ping-pong bf16 [2][B][D]
//   +272695296 : barrier counters (2 x u32)
// ---------------------------------------------------------------------------
extern "C" void kernel_launch(void* const* d_in, const int* in_sizes, int n_in,
                              void* d_out, int out_size, void* d_ws, size_t ws_size,
                              hipStream_t stream)
{
    const float* x   = (const float*)d_in[0];
    const float* h0  = (const float*)d_in[1];
    const float* Wih = (const float*)d_in[2];
    const float* Whh = (const float*)d_in[3];
    const float* bih = (const float*)d_in[4];
    const float* bhh = (const float*)d_in[5];
    float* out = (float*)d_out;

    char* ws = (char*)d_ws;
    float*    XPROJ = (float*)(ws);
    __bf16*   WHHB  = (__bf16*)(ws + 268435456ull);
    __bf16*   WIHB  = (__bf16*)(ws + 270532608ull);
    __bf16*   HBF   = (__bf16*)(ws + 272629760ull);
    unsigned* BAR   = (unsigned*)(ws + 272695296ull);

    // 2*1M weight elems + 16K h0 elems
    k_init<<<(2097152 + 16384 + 255) / 256, 256, 0, stream>>>(
        Wih, Whh, h0, WIHB, WHHB, HBF, BAR);

    // (65536/64) M-blocks * (1024/128) N-blocks = 8192 blocks
    k_xproj<<<8192, 256, 0, stream>>>(x, WIHB, bih, XPROJ);

    // persistent sequential scan
    k_scan<<<8, 256, 0, stream>>>(WHHB, XPROJ, bhh, HBF, out, BAR);
}
